// SparseMultiHeadAttention_12369505812984
// MI455X (gfx1250) — compile-verified, matrix operand order checked by reading, not run on hardware
//
#include <hip/hip_runtime.h>
#include <hip/hip_bf16.h>

// ---------------------------------------------------------------------------
// Sparse (banded) multi-head attention for MI455X / gfx1250.
//   N=4, S=2048, EMBED=1024, HEADS=16, HD=64, band |i-j|<=5.
// Dominant cost: 4 dense GEMMs (8192x1024x1024) -> v_wmma_f32_16x16x32_bf16,
// fed by double-buffered GLOBAL_LOAD_ASYNC_TO_LDS_B128 (ASYNCcnt) DMA.
// Banded attention (~1% of FLOPs) done with per-row VALU online softmax.
// ---------------------------------------------------------------------------

#define EMBED 1024
#define HEADS 16
#define HD    64
#define SPAR  5
#define BATCH 4
#define SEQ   2048

typedef __attribute__((ext_vector_type(16))) __bf16 v16bf;
typedef __attribute__((ext_vector_type(8)))  __bf16 bf16x8;
typedef __attribute__((ext_vector_type(8)))  float  v8f;

union FragBF { uint4 u[2]; v16bf v; };

// Low 32 bits of a generic pointer to __shared__ = LDS byte offset.
__device__ __forceinline__ unsigned lds_off(const void* p) {
    return (unsigned)(size_t)p;
}

// CDNA5 async DMA: global -> LDS, 16B per lane, tracked by ASYNCcnt.
__device__ __forceinline__ void async_ld_b128(unsigned lds_dst, const void* gaddr) {
    asm volatile("global_load_async_to_lds_b128 %0, %1, off"
                 :: "v"(lds_dst), "v"(gaddr) : "memory");
}
__device__ __forceinline__ void wait_async0() {
    asm volatile("s_wait_asynccnt 0x0" ::: "memory");
}

// ------------------------- fp32 -> bf16 convert ----------------------------
__global__ __launch_bounds__(256)
void cvt_f32_bf16(const float* __restrict__ in, __bf16* __restrict__ out, int n8)
{
    int i      = blockIdx.x * blockDim.x + threadIdx.x;
    int stride = gridDim.x * blockDim.x;
    for (; i < n8; i += stride) {
        const float4* p = reinterpret_cast<const float4*>(in) + (size_t)i * 2;
        float4 f0 = p[0];
        float4 f1 = p[1];
        __bf16 r[8] = { (__bf16)f0.x, (__bf16)f0.y, (__bf16)f0.z, (__bf16)f0.w,
                        (__bf16)f1.x, (__bf16)f1.y, (__bf16)f1.z, (__bf16)f1.w };
        reinterpret_cast<uint4*>(out)[i] = *reinterpret_cast<uint4*>(r);
    }
}

// ------------------------- bf16 WMMA GEMM ----------------------------------
// C[M,N] = A[M,K] * W[N,K]^T   (both row-major, K contiguous)
// Block tile: BM=128, BN=256, BK=32. 256 threads = 8 waves in a 2(M) x 4(N)
// grid; each wave computes a 64x64 tile = 4x4 wmma_f32_16x16x32_bf16 accums.
// Tiles staged via double-buffered global_load_async_to_lds_b128.
template <bool F32OUT>
__global__ __launch_bounds__(256)
void gemm_bf16(const __bf16* __restrict__ A,
               const __bf16* __restrict__ W,
               void* __restrict__ Cout,
               const float* __restrict__ bias,
               int M, int N, int K)
{
    constexpr int BM = 128, BN = 256, BK = 32;
    constexpr int LDT   = 40;                    // row pitch (bf16): 80 B, 16B-aligned
    constexpr int ASZ   = BM * LDT;              // A tile, ushort units
    constexpr int BSZ   = BN * LDT;              // B tile
    constexpr int BUFSZ = ASZ + BSZ;             // one double-buffer slice
    __shared__ unsigned short smem[2 * BUFSZ];   // ~60 KB of the 320 KB WGP LDS

    const int tid  = threadIdx.x;
    const int lane = tid & 31;
    const int wave = tid >> 5;
    const int wm   = wave & 1;    // M wave offset: 64*wm
    const int wn   = wave >> 1;   // N wave offset: 64*wn

    const int blockM = blockIdx.y * BM;
    const int blockN = blockIdx.x * BN;

    // -------- async staging map (per thread, per tile) --------
    // A: 128 rows x 64 B -> thread t: row t/2, 2 chunks at byte col (t&1)*32.
    // B: 256 rows x 64 B -> thread t: row t,   4 chunks.
    const int ar = tid >> 1;
    const int ac = (tid & 1) * 16;               // bf16 col offset (2 x 16B chunks)
    const __bf16* gA = A + (size_t)(blockM + ar) * K + ac;
    const __bf16* gW = W + (size_t)(blockN + tid) * K;
    const unsigned ldsA = lds_off(&smem[0]) + (unsigned)(ar * LDT + ac) * 2;
    const unsigned ldsB = lds_off(&smem[ASZ]) + (unsigned)(tid * LDT) * 2;

    v8f acc[4][4];
#pragma unroll
    for (int i = 0; i < 4; ++i)
#pragma unroll
        for (int j = 0; j < 4; ++j)
            acc[i][j] = (v8f){0.f, 0.f, 0.f, 0.f, 0.f, 0.f, 0.f, 0.f};

    // WMMA 16-bit fragment addressing (ISA 7.12.2):
    // lane L holds row M=L%16; K-halves [kb..kb+7], [kb+16..kb+23], kb=8*(L/16)
    const int frow = lane & 15;
    const int fkb  = (lane >> 4) * 8;

    const int nk = K / BK;

    // ---- prologue: DMA tile 0 into buffer 0 ----
    {
        async_ld_b128(ldsA,      gA);
        async_ld_b128(ldsA + 16, gA + 8);
#pragma unroll
        for (int c = 0; c < 4; ++c)
            async_ld_b128(ldsB + c * 16, gW + c * 8);
    }
    wait_async0();
    __syncthreads();

    for (int kt = 0; kt < nk; ++kt) {
        const unsigned bufO = (unsigned)((kt & 1) * BUFSZ);

        // ---- DMA next tile into the other buffer while we compute ----
        if (kt + 1 < nk) {
            const unsigned nbuf = (unsigned)(((kt + 1) & 1) * BUFSZ) * 2; // bytes
            const __bf16* pA = gA + (size_t)(kt + 1) * BK;
            const __bf16* pW = gW + (size_t)(kt + 1) * BK;
            async_ld_b128(ldsA + nbuf,      pA);
            async_ld_b128(ldsA + nbuf + 16, pA + 8);
#pragma unroll
            for (int c = 0; c < 4; ++c)
                async_ld_b128(ldsB + nbuf + c * 16, pW + c * 8);
            if (kt + 2 < nk) {
                __builtin_prefetch(gA + (size_t)(kt + 2) * BK, 0, 1);
                __builtin_prefetch(gW + (size_t)(kt + 2) * BK, 0, 1);
            }
        }

        // ---- compute from current buffer ----
        const unsigned short* Ab = smem + bufO;
        const unsigned short* Bb = Ab + ASZ;

        FragBF fb[4];
#pragma unroll
        for (int tn = 0; tn < 4; ++tn) {
            const int row = wn * 64 + tn * 16 + frow;
            fb[tn].u[0] = *reinterpret_cast<const uint4*>(&Bb[row * LDT + fkb]);
            fb[tn].u[1] = *reinterpret_cast<const uint4*>(&Bb[row * LDT + fkb + 16]);
        }
#pragma unroll
        for (int tm = 0; tm < 4; ++tm) {
            FragBF fa;
            const int row = wm * 64 + tm * 16 + frow;
            fa.u[0] = *reinterpret_cast<const uint4*>(&Ab[row * LDT + fkb]);
            fa.u[1] = *reinterpret_cast<const uint4*>(&Ab[row * LDT + fkb + 16]);
#pragma unroll
            for (int tn = 0; tn < 4; ++tn)
                acc[tm][tn] = __builtin_amdgcn_wmma_f32_16x16x32_bf16(
                    false, fa.v, false, fb[tn].v,
                    (short)0, acc[tm][tn], false, false);
        }

        // my DMA done, then everyone's: next iteration's buffer is ready
        wait_async0();
        __syncthreads();
    }

    // Epilogue. C/D layout: VGPR r, lanes 0-15 -> M=r, lanes 16-31 -> M=r+8; N=lane%16.
#pragma unroll
    for (int tm = 0; tm < 4; ++tm) {
        const int rbase = blockM + wm * 64 + tm * 16 + ((lane >> 4) * 8);
#pragma unroll
        for (int tn = 0; tn < 4; ++tn) {
            const int col = blockN + wn * 64 + tn * 16 + (lane & 15);
#pragma unroll
            for (int r = 0; r < 8; ++r) {
                const float v   = acc[tm][tn][r];
                const size_t ix = (size_t)(rbase + r) * N + col;
                if (F32OUT) {
                    static_cast<float*>(Cout)[ix] = v + bias[col];
                } else {
                    static_cast<__bf16*>(Cout)[ix] = (__bf16)v;
                }
            }
        }
    }
}

// ------------------------- banded attention --------------------------------
// One thread per (n, h, i) query row; online softmax over the 11-wide band.
__global__ __launch_bounds__(256)
void banded_attention(const __bf16* __restrict__ Q,
                      const __bf16* __restrict__ Kt,
                      const __bf16* __restrict__ Vt,
                      const unsigned char* __restrict__ mask, // (N,1,S,S) bool
                      __bf16* __restrict__ O)
{
    const int tid = blockIdx.x * blockDim.x + threadIdx.x;
    const int i = tid % SEQ;
    const int h = (tid / SEQ) % HEADS;
    const int n = tid / (SEQ * HEADS);

    const size_t rowBase = ((size_t)n * SEQ + i) * EMBED + (size_t)h * HD;
    const unsigned char* mrow = mask + ((size_t)n * SEQ + i) * SEQ;

    float q[HD];
#pragma unroll
    for (int c = 0; c < HD / 8; ++c) {
        bf16x8 qv = reinterpret_cast<const bf16x8*>(Q + rowBase)[c];
#pragma unroll
        for (int e = 0; e < 8; ++e) q[c * 8 + e] = (float)qv[e];
    }

    float m = -1e30f, denom = 0.f;
    float acc[HD];
#pragma unroll
    for (int d = 0; d < HD; ++d) acc[d] = 0.f;

#pragma unroll 1
    for (int t = 0; t < 2 * SPAR + 1; ++t) {
        const int j = i - SPAR + t;
        const bool valid = (j >= 0) && (j < SEQ);
        const int jc = valid ? j : i;

        const __bf16* krow = Kt + ((size_t)n * SEQ + jc) * EMBED + (size_t)h * HD;
        float s = 0.f;
#pragma unroll
        for (int c = 0; c < HD / 8; ++c) {
            bf16x8 kv = reinterpret_cast<const bf16x8*>(krow)[c];
#pragma unroll
            for (int e = 0; e < 8; ++e) s += q[c * 8 + e] * (float)kv[e];
        }
        s *= 0.125f; // 1/sqrt(HD)

        const bool ok = valid && (mrow[jc] != 0);
        float corr, p, mnew;
        if (ok) {
            mnew = fmaxf(m, s);
            corr = __expf(m - mnew);
            p    = __expf(s - mnew);
        } else {
            mnew = m; corr = 1.f; p = 0.f;
        }

        const __bf16* vrow = Vt + ((size_t)n * SEQ + jc) * EMBED + (size_t)h * HD;
        denom = denom * corr + p;
#pragma unroll
        for (int c = 0; c < HD / 8; ++c) {
            bf16x8 vv = reinterpret_cast<const bf16x8*>(vrow)[c];
#pragma unroll
            for (int e = 0; e < 8; ++e)
                acc[c * 8 + e] = acc[c * 8 + e] * corr + p * (float)vv[e];
        }
        m = mnew;
    }

    const float inv = 1.f / denom;
#pragma unroll
    for (int c = 0; c < HD / 8; ++c) {
        bf16x8 ov;
#pragma unroll
        for (int e = 0; e < 8; ++e) ov[e] = (__bf16)(acc[c * 8 + e] * inv);
        reinterpret_cast<bf16x8*>(O + rowBase)[c] = ov;
    }
}

// ------------------------------ launcher -----------------------------------
extern "C" void kernel_launch(void* const* d_in, const int* in_sizes, int n_in,
                              void* d_out, int out_size, void* d_ws, size_t ws_size,
                              hipStream_t stream)
{
    (void)in_sizes; (void)n_in; (void)out_size; (void)ws_size;

    // setup_inputs() order:
    const float* values = (const float*)d_in[0];
    const float* keys   = (const float*)d_in[1];
    const float* query  = (const float*)d_in[2];
    const unsigned char* mask = (const unsigned char*)d_in[3]; // bool, 1B
    const float* Wv = (const float*)d_in[4];
    const float* Wk = (const float*)d_in[5];
    const float* Wq = (const float*)d_in[6];
    const float* Wo = (const float*)d_in[7];
    const float* bo = (const float*)d_in[8];
    float* out = (float*)d_out;

    const size_t SE = (size_t)BATCH * SEQ * EMBED; // 8388608
    const size_t EE = (size_t)EMBED * EMBED;       // 1048576
    const int M = BATCH * SEQ;                     // 8192

    // workspace layout (bf16) ~109 MB total
    __bf16* Xv  = (__bf16*)d_ws;
    __bf16* Xk  = Xv + SE;
    __bf16* Xq  = Xk + SE;
    __bf16* Wvb = Xq + SE;
    __bf16* Wkb = Wvb + EE;
    __bf16* Wqb = Wkb + EE;
    __bf16* Wob = Wqb + EE;
    __bf16* Qb  = Wob + EE;
    __bf16* Kb  = Qb + SE;
    __bf16* Vb  = Kb + SE;
    __bf16* Ob  = Xv; // reuse: Xv consumed by the V projection before attention

    // 1) fp32 -> bf16 conversions
    {
        const int n8a = (int)(SE / 8), n8w = (int)(EE / 8);
        const int ba = min((n8a + 255) / 256, 4096);
        const int bw = min((n8w + 255) / 256, 4096);
        cvt_f32_bf16<<<ba, 256, 0, stream>>>(values, Xv, n8a);
        cvt_f32_bf16<<<ba, 256, 0, stream>>>(keys,   Xk, n8a);
        cvt_f32_bf16<<<ba, 256, 0, stream>>>(query,  Xq, n8a);
        cvt_f32_bf16<<<bw, 256, 0, stream>>>(Wv, Wvb, n8w);
        cvt_f32_bf16<<<bw, 256, 0, stream>>>(Wk, Wkb, n8w);
        cvt_f32_bf16<<<bw, 256, 0, stream>>>(Wq, Wqb, n8w);
        cvt_f32_bf16<<<bw, 256, 0, stream>>>(Wo, Wob, n8w);
    }

    // 2) QKV projections: X @ W^T  (WMMA bf16, async-LDS pipelined)
    dim3 ggrid(EMBED / 256, M / 128); // (4, 64)
    gemm_bf16<false><<<ggrid, 256, 0, stream>>>(Xq, Wqb, Qb, nullptr, M, EMBED, EMBED);
    gemm_bf16<false><<<ggrid, 256, 0, stream>>>(Xk, Wkb, Kb, nullptr, M, EMBED, EMBED);
    gemm_bf16<false><<<ggrid, 256, 0, stream>>>(Xv, Wvb, Vb, nullptr, M, EMBED, EMBED);

    // 3) banded softmax attention
    {
        const int rows = BATCH * HEADS * SEQ; // 131072
        banded_attention<<<rows / 256, 256, 0, stream>>>(Qb, Kb, Vb, mask, Ob);
    }

    // 4) output projection + bias (f32 out)
    gemm_bf16<true><<<ggrid, 256, 0, stream>>>(Ob, Wob, out, bo, M, EMBED, EMBED);
}
